// AttentionMemory_35433480192809
// MI455X (gfx1250) — compile-verified, matrix operand order checked by reading, not run on hardware
//
#include <hip/hip_runtime.h>
#include <hip/hip_bf16.h>
#include <float.h>

#define BATCH 16
#define CK    64
#define HW    2304        // 36*64
#define MTILE 144         // HW/16 m-tiles
#define QPW   16          // queries per wave
#define WAVES 8
#define QPB   (QPW*WAVES) // 128 queries per workgroup
#define QBLK  (HW/QPB)    // 18 workgroups along q
#define TILE_BYTES 2048   // one 16m x 64k bf16 tile in fragment layout

// scale = (1/sqrt(CK)) * log2(e)  -> logits expressed in base-2
#define S2      0.18033688011112042f   // 0.125 * 1.4426950408889634
#define INV_S2  5.545177444479562f     // 8 * ln(2)

typedef __attribute__((ext_vector_type(16))) __bf16 bf16x16;
typedef __attribute__((ext_vector_type(2)))  __bf16 bf16x2;
typedef __attribute__((ext_vector_type(8)))  float  f32x8;

// swap data between lane n and lane n+16 (wave32): ds_swizzle SWAPX16
__device__ __forceinline__ float lane_swap16(float x) {
    return __int_as_float(__builtin_amdgcn_ds_swizzle(__float_as_int(x), 0x401F));
}

// Load one bf16 B fragment column for this lane straight from global f32.
__device__ __forceinline__ bf16x16 load_bfrag(const float* __restrict__ col, int kadd) {
    bf16x16 f;
#pragma unroll
    for (int e = 0; e < 16; ++e) {
        const int k = ((e >> 3) << 4) + kadd + (e & 7);
        f[e] = (__bf16)col[(size_t)k * HW];
    }
    return f;
}

__global__ __launch_bounds__(256)
void attn_affinity_softmax(const float* __restrict__ mk,
                           const float* __restrict__ qk,
                           float* __restrict__ out) {
    // Entire mk[b] slice as bf16 in WMMA A-fragment layout: 144 tiles x 2KB = 288KB.
    __shared__ __align__(32) unsigned int frag_lds[(MTILE * TILE_BYTES) / 4];
    __shared__ float a_lds[HW];   // per-row ||mk[:,m]||^2

    const int b    = blockIdx.y;
    const int t    = threadIdx.x;
    const int wave = t >> 5;
    const int lane = t & 31;
    const int half = lane >> 4;
    const int nr   = lane & 15;
    const int kadd = half * 8;

    const float* __restrict__ mkb = mk + (size_t)b * CK * HW;
    const float* __restrict__ qkb = qk + (size_t)b * CK * HW;

    // ---- row norms a[m] = sum_k mk[k,m]^2 (coalesced; runs once) ----
#pragma unroll 1
    for (int i = 0; i < HW / 256; ++i) {
        const int m = t + i * 256;
        const float* p = mkb + m;
        float s = 0.f;
#pragma unroll
        for (int k = 0; k < CK; ++k) { const float v = p[(size_t)k * HW]; s += v * v; }
        a_lds[m] = s;
    }

    // ---- stage mk[b] -> LDS as bf16 fragments (pairs of consecutive k) ----
    // mapping: frag=k>=32; k'=k&31; fh=(k'>>3)&1; e=((k'>>4)<<3)|(k'&7); lane=(m&15)+16*fh
#pragma unroll 1
    for (int k2 = 0; k2 < CK / 2; ++k2) {
        const int k    = 2 * k2;
        const int frag = (k >= 32) ? 1 : 0;
        const int kp   = k & 31;
        const int fh   = (kp >> 3) & 1;
        const int e    = ((kp >> 4) << 3) | (kp & 7);     // even
        const float* r0 = mkb + (size_t)k * HW;
        const float* r1 = mkb + (size_t)(k + 1) * HW;
#pragma unroll
        for (int j = 0; j < HW / 256; ++j) {
            const int m  = t + j * 256;
            const int mt = m >> 4, mr = m & 15;
            bf16x2 p2;
            p2[0] = (__bf16)r0[m];
            p2[1] = (__bf16)r1[m];
            const int baddr = mt * TILE_BYTES + frag * 1024 + (mr + 16 * fh) * 32 + e * 2;
            frag_lds[baddr >> 2] = __builtin_bit_cast(unsigned int, p2);
        }
    }
    __syncthreads();

    // this lane's query column
    const int q = blockIdx.x * QPB + wave * QPW + nr;

    // c[q] = sum_k qk[k,q]^2  (guarantees logits <= ~0: -||mk-qk||^2 / sqrt(CK))
    float cq = 0.f;
    {
        const float* p = qkb + q;
#pragma unroll
        for (int k = 0; k < CK; ++k) { const float v = p[(size_t)k * HW]; cq += v * v; }
    }

    // B fragments (qk tile), kept live across both passes
    const float* qcol = qkb + q;
    const bf16x16 Blo = load_bfrag(qcol, kadd);
    const bf16x16 Bhi = load_bfrag(qcol, kadd + 32);

    const char* myfrag = (const char*)frag_lds + lane * 32;

    // ---- pass 1: denominator sum (no max subtraction needed: logits <= 0) ----
    float rsum = 0.f;
    for (int mt = 0; mt < MTILE; ++mt) {
        const char* tp = myfrag + mt * TILE_BYTES;
        const bf16x16 Alo = *(const bf16x16*)tp;
        const bf16x16 Ahi = *(const bf16x16*)(tp + 1024);
        f32x8 acc = {};
        acc = __builtin_amdgcn_wmma_f32_16x16x32_bf16(false, Alo, false, Blo,
                                                      (short)0, acc, false, false);
        acc = __builtin_amdgcn_wmma_f32_16x16x32_bf16(false, Ahi, false, Bhi,
                                                      (short)0, acc, false, false);
#pragma unroll
        for (int r = 0; r < 8; ++r) {
            const int m = mt * 16 + r + 8 * half;
            const float l2 = (2.f * acc[r] - a_lds[m] - cq) * S2;   // base-2 logit
            rsum += __builtin_amdgcn_exp2f(l2);
        }
    }

    // merge the two half-wave partials (lanes n and n+16 share column q),
    // then fold normalization into the exponent: p = 2^(l2 - log2(sum))
    const float gsum = rsum + lane_swap16(rsum);
    const float cqL  = cq + __builtin_amdgcn_logf(gsum) * INV_S2;

    // ---- pass 2: recompute, normalize via exponent bias, single write ----
    float* __restrict__ ocol = out + (size_t)b * HW * HW + q;
    for (int mt = 0; mt < MTILE; ++mt) {
        const char* tp = myfrag + mt * TILE_BYTES;
        const bf16x16 Alo = *(const bf16x16*)tp;
        const bf16x16 Ahi = *(const bf16x16*)(tp + 1024);
        f32x8 acc = {};
        acc = __builtin_amdgcn_wmma_f32_16x16x32_bf16(false, Alo, false, Blo,
                                                      (short)0, acc, false, false);
        acc = __builtin_amdgcn_wmma_f32_16x16x32_bf16(false, Ahi, false, Bhi,
                                                      (short)0, acc, false, false);
        float* orow = ocol + (size_t)(mt * 16) * HW;
#pragma unroll
        for (int r = 0; r < 8; ++r) {
            const int mloc = r + 8 * half;
            const int m    = mt * 16 + mloc;
            const float l2 = (2.f * acc[r] - a_lds[m] - cqL) * S2;
            orow[(size_t)mloc * HW] = __builtin_amdgcn_exp2f(l2);
        }
    }
}

extern "C" void kernel_launch(void* const* d_in, const int* in_sizes, int n_in,
                              void* d_out, int out_size, void* d_ws, size_t ws_size,
                              hipStream_t stream) {
    (void)in_sizes; (void)n_in; (void)out_size; (void)d_ws; (void)ws_size;
    const float* mk = (const float*)d_in[0];
    const float* qk = (const float*)d_in[1];
    float* out = (float*)d_out;
    dim3 grid(QBLK, BATCH);   // 18 x 16 workgroups, 8 waves each
    attn_affinity_softmax<<<grid, 256, 0, stream>>>(mk, qk, out);
}
